// Decoder_26293789786523
// MI455X (gfx1250) — compile-verified, matrix operand order checked by reading, not run on hardware
//
#include <hip/hip_runtime.h>
#include <math.h>

// ---------------------------------------------------------------------------
// Shapes (fixed by the reference):
//   S=64, T=64, B=32, E=512, H=512, O=512
// ---------------------------------------------------------------------------

typedef __bf16 bf16_t;
typedef __attribute__((ext_vector_type(16))) __bf16 v16bf;
typedef __attribute__((ext_vector_type(8)))  float  v8f;

union FragU {
    v16bf v;
    uint4 q[2];
};

// Load one 16x32 bf16 fragment (A: rows=M, or B: rows=N of row-major W[N,K]).
// Per CDNA5 ISA 16-bit layout: lane L holds row (L&15); lanes 0-15 carry
// K = {k0..k0+7, k0+16..k0+23}, lanes 16-31 carry K = {k0+8.., k0+24..}.
// Two global_load_b128 per lane.
__device__ __forceinline__ v16bf load_frag(const bf16_t* base, int ld,
                                           int row0, int k0, int lane) {
    int r  = row0 + (lane & 15);
    int hi = (lane >> 4) & 1;
    const bf16_t* p = base + (size_t)r * ld + (k0 + hi * 8);
    FragU f;
    f.q[0] = *(const uint4*)(p);        // K chunk [k0+hi*8     .. +7]
    f.q[1] = *(const uint4*)(p + 16);   // K chunk [k0+16+hi*8  .. +7]
    return f.v;
}

__device__ __forceinline__ float sigm(float x) { return 1.f / (1.f + expf(-x)); }

// ---------------------------------------------------------------------------
// Fused LSTM cell: gates GEMM (up to 3 K-segments, X @ W^T layout) + LSTM
// pointwise, all in one kernel.  Each wave computes the four gate tiles
// (i,f,g,o) = N columns {n, n+512, n+1024, n+1536} for one 16x16 (m,n) block;
// the four f32 accumulators are lane-aligned so the nonlinearity fuses into
// the epilogue:  c' = sig(f)*c + sig(i)*tanh(g);  h' = sig(o)*tanh(c').
// All 5 fragment loads (1xA + 4xB) are issued before the 4 WMMAs so the
// loads clause together and the waits stay partial.
// Tiles: Mtiles=2 (B=32) x Nq=32 (H=512/16) -> 64 waves -> grid 16 x 128thr.
// ---------------------------------------------------------------------------
#define GATE_SEG(Ax, ldax, Bx, ldbx, Kx)                                           \
    for (int k = 0; k < (Kx); k += 32) {                                           \
        v16bf a  = load_frag((Ax), (ldax), mrow, k, lane);                         \
        v16bf b0 = load_frag((Bx), (ldbx), nrow,        k, lane);                  \
        v16bf b1 = load_frag((Bx), (ldbx), nrow + 512,  k, lane);                  \
        v16bf b2 = load_frag((Bx), (ldbx), nrow + 1024, k, lane);                  \
        v16bf b3 = load_frag((Bx), (ldbx), nrow + 1536, k, lane);                  \
        acc[0] = __builtin_amdgcn_wmma_f32_16x16x32_bf16(false, a, false, b0,      \
                                                         (short)0, acc[0], false, false); \
        acc[1] = __builtin_amdgcn_wmma_f32_16x16x32_bf16(false, a, false, b1,      \
                                                         (short)0, acc[1], false, false); \
        acc[2] = __builtin_amdgcn_wmma_f32_16x16x32_bf16(false, a, false, b2,      \
                                                         (short)0, acc[2], false, false); \
        acc[3] = __builtin_amdgcn_wmma_f32_16x16x32_bf16(false, a, false, b3,      \
                                                         (short)0, acc[3], false, false); \
    }

__global__ void lstm_gemm_k(const bf16_t* A0, int lda0, const bf16_t* B0, int ldb0, int K0,
                            const bf16_t* A1, int lda1, const bf16_t* B1, int ldb1, int K1,
                            const bf16_t* A2, int lda2, const bf16_t* B2, int ldb2, int K2,
                            const float* bias,       // [2048] = bih + bhh
                            float* c,                // [32][512] in/out
                            bf16_t* hb,              // [32][512] out (bf16)
                            float* hf) {             // [32][512] out (f32) or null
    int wave = threadIdx.x >> 5;
    int lane = threadIdx.x & 31;
    int tile = blockIdx.x * (blockDim.x >> 5) + wave;
    if (tile >= 2 * 32) return;            // wave-uniform
    int nq = tile & 31;
    int mt = tile >> 5;
    int mrow = mt << 4;
    int nrow = nq << 4;

    v8f acc[4];
#pragma unroll
    for (int g = 0; g < 4; g++) acc[g] = (v8f){0.f,0.f,0.f,0.f,0.f,0.f,0.f,0.f};

    GATE_SEG(A0, lda0, B0, ldb0, K0)
    GATE_SEG(A1, lda1, B1, ldb1, K1)
    GATE_SEG(A2, lda2, B2, ldb2, K2)

    // C/D layout: lanes 0-15 vgpr i -> M=i ; lanes 16-31 vgpr i -> M=8+i ; N=lane&15
    int m0 = mrow + ((lane >> 4) << 3);
    int n  = nrow + (lane & 15);
    float b_i = bias[n];
    float b_f = bias[n + 512];
    float b_g = bias[n + 1024];
    float b_o = bias[n + 1536];
#pragma unroll
    for (int i = 0; i < 8; i++) {
        int idx  = (m0 + i) * 512 + n;
        float iv = sigm(acc[0][i] + b_i);
        float fv = sigm(acc[1][i] + b_f);
        float gv = tanhf(acc[2][i] + b_g);
        float ov = sigm(acc[3][i] + b_o);
        float c2 = fv * c[idx] + iv * gv;
        float h2 = ov * tanhf(c2);
        c[idx]  = c2;
        hb[idx] = (bf16_t)h2;
        if (hf) hf[idx] = h2;
    }
}

// ---------------------------------------------------------------------------
// Generic bf16 WMMA GEMM for the remaining projections.
// K manually unrolled x2 so two A/B fragment pairs are in flight before the
// first WMMA of each iteration.
// mode 1: tanh -> bf16 store row-major [M][N]         (attention output / feed)
// mode 2: f32 + bias[N], scatter out[b*O*T + n*T + t] with m = t*32+b (pred)
// ---------------------------------------------------------------------------
#define GEMM_SEG(Ax, ldax, Bx, ldbx, Kx)                                           \
    for (int k = 0; k < (Kx); k += 64) {                                           \
        v16bf a0 = load_frag((Ax), (ldax), mrow, k,      lane);                    \
        v16bf b0 = load_frag((Bx), (ldbx), nrow, k,      lane);                    \
        v16bf a1 = load_frag((Ax), (ldax), mrow, k + 32, lane);                    \
        v16bf b1 = load_frag((Bx), (ldbx), nrow, k + 32, lane);                    \
        acc = __builtin_amdgcn_wmma_f32_16x16x32_bf16(false, a0, false, b0,        \
                                                      (short)0, acc, false, false);\
        acc = __builtin_amdgcn_wmma_f32_16x16x32_bf16(false, a1, false, b1,        \
                                                      (short)0, acc, false, false);\
    }

__global__ void gemm_wmma_k(const bf16_t* A0, int lda0, const bf16_t* B0, int ldb0, int K0,
                            const bf16_t* A1, int lda1, const bf16_t* B1, int ldb1, int K1,
                            const float* bias, void* out,
                            int Mtiles, int Ntiles, int mode) {
    int wave = threadIdx.x >> 5;
    int lane = threadIdx.x & 31;
    int tile = blockIdx.x * (blockDim.x >> 5) + wave;
    if (tile >= Mtiles * Ntiles) return;   // wave-uniform: EXEC stays all-ones
    int nt = tile % Ntiles;
    int mt = tile / Ntiles;
    int mrow = mt << 4;
    int nrow = nt << 4;

    v8f acc = {0.f, 0.f, 0.f, 0.f, 0.f, 0.f, 0.f, 0.f};

    GEMM_SEG(A0, lda0, B0, ldb0, K0)
    GEMM_SEG(A1, lda1, B1, ldb1, K1)

    int M0    = mrow + ((lane >> 4) << 3);
    int N     = nrow + (lane & 15);
    int ldout = Ntiles << 4;

    if (mode == 1) {
        bf16_t* o = (bf16_t*)out;
#pragma unroll
        for (int i = 0; i < 8; i++)
            o[(size_t)(M0 + i) * ldout + N] = (bf16_t)tanhf(acc[i]);
    } else {
        float* o = (float*)out;
        float bv = bias ? bias[N] : 0.f;
#pragma unroll
        for (int i = 0; i < 8; i++) {
            int m = M0 + i;
            int b = m & 31;      // m = t*B + b, B = 32
            int t = m >> 5;
            o[(size_t)b * (512 * 64) + (size_t)N * 64 + t] = acc[i] + bv;
        }
    }
}

// ---------------------------------------------------------------------------
// Small helper kernels
// ---------------------------------------------------------------------------
__global__ void f2bf_k(const float* in, bf16_t* out, int n) {
    int i = blockIdx.x * 256 + threadIdx.x;
    if (i < n) out[i] = (bf16_t)in[i];
}

__global__ void addbias_k(const float* a, const float* b, float* o, int n) {
    int i = blockIdx.x * 256 + threadIdx.x;
    if (i < n) o[i] = a[i] + b[i];
}

// dst_emb[t][b][e] = embed[dst[t*B+b]][e]  (converted to bf16)
__global__ void gather_embed_k(const int* dst, const float* embed, bf16_t* out) {
    int i  = blockIdx.x * 256 + threadIdx.x;     // < T*B*E = 64*32*512
    int tb = i >> 9;
    int e  = i & 511;
    out[i] = (bf16_t)embed[(size_t)dst[tb] * 512 + e];
}

__global__ void init_state_k(const float* enc_hid, const float* enc_mem,
                             bf16_t* h0b, bf16_t* h1b,
                             float* c0, float* c1, bf16_t* feed0) {
    int i = blockIdx.x * 256 + threadIdx.x;      // < B*H = 16384
    h0b[i]   = (bf16_t)enc_hid[i];
    h1b[i]   = (bf16_t)enc_hid[16384 + i];
    c0[i]    = enc_mem[i];
    c1[i]    = enc_mem[16384 + i];
    feed0[i] = (bf16_t)0.0f;
}

// Dot-product attention: one block per batch element b.
__global__ void attn_k(const float* enc_out, const unsigned char* mask,
                       const float* h1f, bf16_t* ctxb) {
    int b    = blockIdx.x;
    int tid  = threadIdx.x;                      // 256 threads = 8 waves
    int wave = tid >> 5;
    int lane = tid & 31;

    __shared__ float sh_h[512];
    __shared__ float sc[64];
    __shared__ float sinv;

    sh_h[tid]       = h1f[(size_t)b * 512 + tid];
    sh_h[tid + 256] = h1f[(size_t)b * 512 + 256 + tid];
    __syncthreads();

    for (int s = wave; s < 64; s += 8) {
        const float* e = enc_out + (size_t)s * (32 * 512) + (size_t)b * 512;
        float acc = 0.f;
        for (int k = lane; k < 512; k += 32) acc += e[k] * sh_h[k];
        for (int off = 16; off > 0; off >>= 1) acc += __shfl_down(acc, off, 32);
        if (lane == 0) sc[s] = mask[s * 32 + b] ? -3.0e38f : acc;
    }
    __syncthreads();

    if (tid == 0) {
        float m = -3.0e38f;
        for (int s = 0; s < 64; s++) m = fmaxf(m, sc[s]);
        float sum = 0.f;
        for (int s = 0; s < 64; s++) { float e = expf(sc[s] - m); sc[s] = e; sum += e; }
        sinv = 1.f / sum;
    }
    __syncthreads();
    float inv = sinv;

    for (int h = tid; h < 512; h += 256) {
        float acc = 0.f;
        for (int s = 0; s < 64; s++)
            acc += sc[s] * enc_out[(size_t)s * (32 * 512) + (size_t)b * 512 + h];
        ctxb[(size_t)b * 512 + h] = (bf16_t)(acc * inv);
    }
}

// ---------------------------------------------------------------------------
// Host side
// ---------------------------------------------------------------------------
extern "C" void kernel_launch(void* const* d_in, const int* in_sizes, int n_in,
                              void* d_out, int out_size, void* d_ws, size_t ws_size,
                              hipStream_t stream) {
    (void)in_sizes; (void)n_in; (void)out_size; (void)ws_size;

    const float*         enc_out = (const float*)d_in[0];
    const float*         enc_hid = (const float*)d_in[1];
    const float*         enc_mem = (const float*)d_in[2];
    const unsigned char* mask    = (const unsigned char*)d_in[3];
    const int*           dst     = (const int*)d_in[4];
    const float*         embed   = (const float*)d_in[5];
    const float*         Wih0    = (const float*)d_in[6];
    const float*         Whh0    = (const float*)d_in[7];
    const float*         bih0    = (const float*)d_in[8];
    const float*         bhh0    = (const float*)d_in[9];
    const float*         Wih1    = (const float*)d_in[10];
    const float*         Whh1    = (const float*)d_in[11];
    const float*         bih1    = (const float*)d_in[12];
    const float*         bhh1    = (const float*)d_in[13];
    const float*         Wattn   = (const float*)d_in[14];
    const float*         Wpred   = (const float*)d_in[15];
    const float*         bpred   = (const float*)d_in[16];

    char*  ws  = (char*)d_ws;
    size_t off = 0;
    auto alloc = [&](size_t bytes) -> char* {
        char* p = ws + off;
        off += (bytes + 255) & ~(size_t)255;
        return p;
    };

    bf16_t* Wih0b  = (bf16_t*)alloc((size_t)2048 * 1024 * 2);
    bf16_t* Whh0b  = (bf16_t*)alloc((size_t)2048 * 512 * 2);
    bf16_t* Wih1b  = (bf16_t*)alloc((size_t)2048 * 512 * 2);
    bf16_t* Whh1b  = (bf16_t*)alloc((size_t)2048 * 512 * 2);
    bf16_t* Wattnb = (bf16_t*)alloc((size_t)512 * 1024 * 2);
    bf16_t* Wpredb = (bf16_t*)alloc((size_t)512 * 512 * 2);
    bf16_t* xembb  = (bf16_t*)alloc((size_t)64 * 32 * 512 * 2);
    bf16_t* outsb  = (bf16_t*)alloc((size_t)65 * 32 * 512 * 2); // slot 0 = zero feed
    bf16_t* h0b    = (bf16_t*)alloc((size_t)32 * 512 * 2);
    bf16_t* h1b    = (bf16_t*)alloc((size_t)32 * 512 * 2);
    bf16_t* ctxb   = (bf16_t*)alloc((size_t)32 * 512 * 2);
    float*  c0     = (float*)alloc((size_t)32 * 512 * 4);
    float*  c1     = (float*)alloc((size_t)32 * 512 * 4);
    float*  h1f    = (float*)alloc((size_t)32 * 512 * 4);
    float*  bsum0  = (float*)alloc((size_t)2048 * 4);
    float*  bsum1  = (float*)alloc((size_t)2048 * 4);

    // --- setup: weight conversion, bias fold, embedding gather, state init ---
    auto cvt = [&](const float* src, bf16_t* dstp, int n) {
        f2bf_k<<<(n + 255) / 256, 256, 0, stream>>>(src, dstp, n);
    };
    cvt(Wih0,  Wih0b,  2048 * 1024);
    cvt(Whh0,  Whh0b,  2048 * 512);
    cvt(Wih1,  Wih1b,  2048 * 512);
    cvt(Whh1,  Whh1b,  2048 * 512);
    cvt(Wattn, Wattnb, 512 * 1024);
    cvt(Wpred, Wpredb, 512 * 512);
    addbias_k<<<8, 256, 0, stream>>>(bih0, bhh0, bsum0, 2048);
    addbias_k<<<8, 256, 0, stream>>>(bih1, bhh1, bsum1, 2048);
    gather_embed_k<<<(64 * 32 * 512) / 256, 256, 0, stream>>>(dst, embed, xembb);
    init_state_k<<<64, 256, 0, stream>>>(enc_hid, enc_mem, h0b, h1b, c0, c1, outsb);

    // --- sequential decode over T = 64 steps ---
    for (int t = 0; t < 64; t++) {
        const bf16_t* embt = xembb + (size_t)t * 32 * 512;
        const bf16_t* feed = outsb + (size_t)t * 32 * 512;   // out of step t-1 (slot0 = zeros)

        // Layer 0: gates = emb@Wih0[:,:512]^T + feed@Wih0[:,512:]^T + h0@Whh0^T
        //          fused with LSTM pointwise -> h0b (bf16), c0
        lstm_gemm_k<<<16, 128, 0, stream>>>(
            embt, 512, Wih0b,       1024, 512,
            feed, 512, Wih0b + 512, 1024, 512,
            h0b,  512, Whh0b,        512, 512,
            bsum0, c0, h0b, nullptr);

        // Layer 1: gates = h0@Wih1^T + h1@Whh1^T, fused pointwise -> h1b, h1f, c1
        lstm_gemm_k<<<16, 128, 0, stream>>>(
            h0b, 512, Wih1b, 512, 512,
            h1b, 512, Whh1b, 512, 512,
            h0b, 512, Wih1b, 512, 0,
            bsum1, c1, h1b, h1f);

        // attention over S=64 (VALU; tiny)
        attn_k<<<32, 256, 0, stream>>>(enc_out, mask, h1f, ctxb);

        // out = tanh([ctx|h1] @ Wattn^T) -> bf16 feed for next step
        gemm_wmma_k<<<16, 128, 0, stream>>>(
            ctxb, 512, Wattnb,       1024, 512,
            h1b,  512, Wattnb + 512, 1024, 512,
            nullptr, (void*)(outsb + (size_t)(t + 1) * 32 * 512),
            2, 32, /*mode=*/1);
    }

    // --- final projection: out[b,o,t] = outs[t,b,:]·Wpred[o,:] + bpred[o] ---
    gemm_wmma_k<<<1024, 128, 0, stream>>>(
        outsb + 32 * 512, 512, Wpredb, 512, 512,
        nullptr, 0, nullptr, 0, 0,
        bpred, d_out, /*Mtiles=*/128, /*Ntiles=*/32, /*mode=*/2);
}
